// HybridMoEWrapper_14405320311033
// MI455X (gfx1250) — compile-verified, matrix operand order checked by reading, not run on hardware
//
#include <hip/hip_runtime.h>
#include <hip/hip_bf16.h>

// Problem constants (match reference)
#define T_TOK 1024
#define H_DIM 2048
#define E_NUM 32
#define K_TOP 4
#define F_DIM 1408
#define FS_DIM 5632
#define CAP_E 256

typedef _Float16 v16h __attribute__((ext_vector_type(16)));
typedef _Float16 v8h  __attribute__((ext_vector_type(8)));
typedef float    v8f  __attribute__((ext_vector_type(8)));
typedef int      v4i  __attribute__((ext_vector_type(4)));

union V16HU { v16h v; v8h h[2]; };

__device__ __forceinline__ float fast_sigmoid(float x) {
  return 1.0f / (1.0f + __expf(-x));
}

// CDNA5 async global->LDS copy (ASYNCcnt-tracked), with sync fallback.
#if defined(__gfx1250__) && __has_builtin(__builtin_amdgcn_global_load_async_to_lds_b128)
#define ASYNC_LDS 1
using as1_v4i = __attribute__((address_space(1))) v4i;
using as3_v4i = __attribute__((address_space(3))) v4i;
#endif

__device__ __forceinline__ void stage_a16(const _Float16* src, _Float16* ldsDst) {
#if defined(ASYNC_LDS)
  __builtin_amdgcn_global_load_async_to_lds_b128(
      (as1_v4i*)(v4i*)(void*)src,
      (as3_v4i*)(v4i*)(void*)ldsDst, 0, 0);
  __builtin_amdgcn_global_load_async_to_lds_b128(
      (as1_v4i*)(v4i*)(void*)(src + 8),
      (as3_v4i*)(v4i*)(void*)(ldsDst + 8), 0, 0);
#else
  *(v8h*)ldsDst       = *(const v8h*)src;
  *(v8h*)(ldsDst + 8) = *(const v8h*)(src + 8);
#endif
}

__device__ __forceinline__ void wait_async0() {
#if defined(ASYNC_LDS)
#if __has_builtin(__builtin_amdgcn_s_wait_asynccnt)
  __builtin_amdgcn_s_wait_asynccnt(0);
#else
  asm volatile("s_wait_asynccnt 0x0" ::: "memory");
#endif
#endif
}

// pack two f32 -> two f16 in one dword (for b32 LDS stores of k-pairs)
__device__ __forceinline__ unsigned pack_h2(float a, float b) {
  union { _Float16 h[2]; unsigned u; } p;
  p.h[0] = (_Float16)a;
  p.h[1] = (_Float16)b;
  return p.u;
}

// Load one 16x32 f16 WMMA fragment from LDS (row-major, half-stride strideH).
// Layout per CDNA5 ISA 7.12.2 (16-bit A-matrix 16x32; B uses the mirrored
// mapping from an [N][K] transposed tile):
//   lanes 0-15 : row = row0+lane,    K = 0..7  (h[0]) and 16..23 (h[1])
//   lanes 16-31: row = row0+lane-16, K = 8..15 (h[0]) and 24..31 (h[1])
__device__ __forceinline__ v16h load_frag(const _Float16* base, int row0, int strideH) {
  int lane = threadIdx.x & 31;
  int r = row0 + (lane & 15);
  int kb = (lane < 16) ? 0 : 8;
  const _Float16* p = base + r * strideH + kb;
  V16HU f;
  f.h[0] = *(const v8h*)(p);
  f.h[1] = *(const v8h*)(p + 16);
  return f.v;
}

// ---------------------------------------------------------------------------
// 1) Router: one wave (32 lanes) per token. lane == expert id.
// ---------------------------------------------------------------------------
__global__ __launch_bounds__(32)
void router_kernel(const float* __restrict__ x, const float* __restrict__ Wr,
                   const float* __restrict__ Wse,
                   int* __restrict__ topi, float* __restrict__ topw,
                   float* __restrict__ segate) {
  const int t = blockIdx.x;
  const int lane = threadIdx.x;
  const float* xt = x + (long long)t * H_DIM;

  float logit = 0.f;
  for (int h = 0; h < H_DIM; ++h)
    logit = fmaf(xt[h], Wr[h * E_NUM + lane], logit);

  float se = 0.f;
  for (int h = lane; h < H_DIM; h += 32)
    se = fmaf(xt[h], Wse[h], se);

  // softmax over 32 lanes
  float m = logit;
  for (int off = 16; off; off >>= 1) m = fmaxf(m, __shfl_xor(m, off));
  float p = __expf(logit - m);
  float s = p;
  for (int off = 16; off; off >>= 1) s += __shfl_xor(s, off);
  float prob = p / s;
  for (int off = 16; off; off >>= 1) se += __shfl_xor(se, off);

  // iterative top-4 (deterministic tie-break on lowest index)
  float v = prob;
  float val0 = 0.f, val1 = 0.f, val2 = 0.f, val3 = 0.f;
  int   sel0 = 0,   sel1 = 0,   sel2 = 0,   sel3 = 0;
#pragma unroll
  for (int k = 0; k < 4; ++k) {
    float bv = v; int bi = lane;
    for (int off = 16; off; off >>= 1) {
      float ov = __shfl_xor(bv, off);
      int   oi = __shfl_xor(bi, off);
      if (ov > bv || (ov == bv && oi < bi)) { bv = ov; bi = oi; }
    }
    if      (k == 0) { val0 = bv; sel0 = bi; }
    else if (k == 1) { val1 = bv; sel1 = bi; }
    else if (k == 2) { val2 = bv; sel2 = bi; }
    else             { val3 = bv; sel3 = bi; }
    if (lane == bi) v = -1.0f;
  }
  if (lane == 0) {
    float ws = val0 + val1 + val2 + val3;
    topi[t * 4 + 0] = sel0; topw[t * 4 + 0] = val0 / ws;
    topi[t * 4 + 1] = sel1; topw[t * 4 + 1] = val1 / ws;
    topi[t * 4 + 2] = sel2; topw[t * 4 + 2] = val2 / ws;
    topi[t * 4 + 3] = sel3; topw[t * 4 + 3] = val3 / ws;
    segate[t] = fast_sigmoid(se);
  }
}

// ---------------------------------------------------------------------------
// 2) Slot positions: thread e scans all pairs in order -> reference cumsum.
// ---------------------------------------------------------------------------
__global__ __launch_bounds__(32)
void pos_kernel(const int* __restrict__ topi, int* __restrict__ pos) {
  const int e = threadIdx.x;
  int cnt = 0;
  for (int p = 0; p < T_TOK * K_TOP; ++p) {
    int sel = topi[p];               // broadcast read across lanes
    if (sel == e) { pos[p] = cnt; ++cnt; }
  }
}

// 3a) init slot->token map to -1
__global__ void init_src_kernel(int* __restrict__ src) {
  int i = blockIdx.x * blockDim.x + threadIdx.x;
  if (i < E_NUM * CAP_E) src[i] = -1;
}

// 3b) scatter pair -> slot (unique (e,slot) pairs: race-free)
__global__ void fill_src_kernel(const int* __restrict__ topi, const int* __restrict__ pos,
                                int* __restrict__ src) {
  int p = blockIdx.x * blockDim.x + threadIdx.x;
  if (p < T_TOK * K_TOP) {
    int e = topi[p];
    int s = pos[p];
    if (s < CAP_E) src[e * CAP_E + s] = p >> 2;   // token id
  }
}

// 3c) materialize xe rows as f16 (zeros for unused slots)
__global__ __launch_bounds__(256)
void build_xe_kernel(const float* __restrict__ x, const int* __restrict__ src,
                     _Float16* __restrict__ xe) {
  const int row = blockIdx.x;                 // 0 .. E*CAP-1
  const int t = src[row];
  const int c = threadIdx.x * 8;
  v8h o;
  if (t >= 0) {
    const float* px = x + (long long)t * H_DIM + c;
    float4 a = *(const float4*)(px);
    float4 b = *(const float4*)(px + 4);
    o[0] = (_Float16)a.x; o[1] = (_Float16)a.y; o[2] = (_Float16)a.z; o[3] = (_Float16)a.w;
    o[4] = (_Float16)b.x; o[5] = (_Float16)b.y; o[6] = (_Float16)b.z; o[7] = (_Float16)b.w;
  } else {
#pragma unroll
    for (int j = 0; j < 8; ++j) o[j] = (_Float16)0.0f;
  }
  *(v8h*)(xe + (long long)row * H_DIM + c) = o;
}

// generic fp32 -> f16 convert (8 elems/thread)
__global__ __launch_bounds__(256)
void cvt_f32_f16_kernel(const float* __restrict__ in, _Float16* __restrict__ out,
                        long long n) {
  long long i = ((long long)blockIdx.x * blockDim.x + threadIdx.x) * 8;
  if (i + 8 <= n) {
    float4 a = *(const float4*)(in + i);
    float4 b = *(const float4*)(in + i + 4);
    v8h o;
    o[0] = (_Float16)a.x; o[1] = (_Float16)a.y; o[2] = (_Float16)a.z; o[3] = (_Float16)a.w;
    o[4] = (_Float16)b.x; o[5] = (_Float16)b.y; o[6] = (_Float16)b.z; o[7] = (_Float16)b.w;
    *(v8h*)(out + i) = o;
  }
}

// ---------------------------------------------------------------------------
// 4) Fused dual-GEMM + SwiGLU: Out = f16( silu(A*Bg) * (A*Bu) )
//    A: f16 [M,K] row-major (lda = Kdim), Bg/Bu: f32 [K,N] row-major (ldb = N)
//    Block tile 128(M) x 64(N), K pipelined in 32-deep double-buffered LDS.
//    A staged with GLOBAL_LOAD_ASYNC_TO_LDS_B128; B converted f32->f16 in
//    VALU and stored transposed [n][k] with packed b32 stores.
// ---------------------------------------------------------------------------
#define LDS_STRIDE 40            // halves; 80B rows stay 16B-aligned
#define BUF_A (128 * LDS_STRIDE)
#define BUF_B (64 * LDS_STRIDE)

__global__ __launch_bounds__(256)
void swiglu_gemm_kernel(const _Float16* __restrict__ A, const float* __restrict__ Bg,
                        const float* __restrict__ Bu, _Float16* __restrict__ Out,
                        int N, int Kdim,
                        long long strideA, long long strideB, long long strideOut) {
  __shared__ __align__(16) _Float16 ldsA [2 * BUF_A];
  __shared__ __align__(16) _Float16 ldsBg[2 * BUF_B];
  __shared__ __align__(16) _Float16 ldsBu[2 * BUF_B];

  const int tid = threadIdx.x;
  const int e = blockIdx.z;
  const long long mBase = (long long)blockIdx.y * 128;
  const long long nBase = (long long)blockIdx.x * 64;
  const _Float16* Ae  = A  + e * strideA;
  const float*    Bge = Bg + e * strideB;
  const float*    Bue = Bu + e * strideB;

  const int w  = tid >> 5;
  const int wm = (w & 3) * 32;     // wave M origin in tile
  const int wn = (w >> 2) * 32;    // wave N origin in tile

  v8f accg[2][2], accu[2][2];
#pragma unroll
  for (int i = 0; i < 2; ++i)
#pragma unroll
    for (int j = 0; j < 2; ++j)
#pragma unroll
      for (int r = 0; r < 8; ++r) { accg[i][j][r] = 0.f; accu[i][j][r] = 0.f; }

  // staging roles
  const int ar = tid >> 1;          // 0..127 (A row)
  const int ac = (tid & 1) * 16;    // 0 or 16 (A col, halves)
  const int k2 = (tid >> 4) * 2;    // 0..30 even (B k-row pair)
  const int nc = (tid & 15) * 4;    // 0..60 (B n-col group)

  // ---- prologue: stage chunk 0 into buffer 0
  {
    stage_a16(Ae + (mBase + ar) * (long long)Kdim + ac, &ldsA[ar * LDS_STRIDE + ac]);
    const float* pg = Bge + (long long)k2 * N + nBase + nc;
    const float* pu = Bue + (long long)k2 * N + nBase + nc;
    float4 g0 = *(const float4*)(pg);
    float4 g1 = *(const float4*)(pg + N);
    float4 u0 = *(const float4*)(pu);
    float4 u1 = *(const float4*)(pu + N);
    float g0v[4] = {g0.x, g0.y, g0.z, g0.w}, g1v[4] = {g1.x, g1.y, g1.z, g1.w};
    float u0v[4] = {u0.x, u0.y, u0.z, u0.w}, u1v[4] = {u1.x, u1.y, u1.z, u1.w};
#pragma unroll
    for (int j = 0; j < 4; ++j) {
      *(unsigned*)&ldsBg[(nc + j) * LDS_STRIDE + k2] = pack_h2(g0v[j], g1v[j]);
      *(unsigned*)&ldsBu[(nc + j) * LDS_STRIDE + k2] = pack_h2(u0v[j], u1v[j]);
    }
    wait_async0();
  }
  __syncthreads();

  int buf = 0;
  for (int kb = 0; kb < Kdim; kb += 32) {
    const bool has_next = (kb + 32) < Kdim;
    _Float16* A_cur  = ldsA  + buf * BUF_A;
    _Float16* Bg_cur = ldsBg + buf * BUF_B;
    _Float16* Bu_cur = ldsBu + buf * BUF_B;
    _Float16* A_nxt  = ldsA  + (buf ^ 1) * BUF_A;
    _Float16* Bg_nxt = ldsBg + (buf ^ 1) * BUF_B;
    _Float16* Bu_nxt = ldsBu + (buf ^ 1) * BUF_B;

    float4 g0{}, g1{}, u0{}, u1{};
    if (has_next) {
      // issue next-chunk loads early: async A -> LDS, B -> VGPRs
      stage_a16(Ae + (mBase + ar) * (long long)Kdim + kb + 32 + ac,
                &A_nxt[ar * LDS_STRIDE + ac]);
      const float* pg = Bge + (long long)(kb + 32 + k2) * N + nBase + nc;
      const float* pu = Bue + (long long)(kb + 32 + k2) * N + nBase + nc;
      g0 = *(const float4*)(pg);
      g1 = *(const float4*)(pg + N);
      u0 = *(const float4*)(pu);
      u1 = *(const float4*)(pu + N);
    }

    // compute on current buffer
    v16h afrag[2], bgf[2], bufg[2];
#pragma unroll
    for (int i = 0; i < 2; ++i) afrag[i] = load_frag(A_cur, wm + i * 16, LDS_STRIDE);
#pragma unroll
    for (int j = 0; j < 2; ++j) {
      bgf[j]  = load_frag(Bg_cur, wn + j * 16, LDS_STRIDE);
      bufg[j] = load_frag(Bu_cur, wn + j * 16, LDS_STRIDE);
    }
#pragma unroll
    for (int i = 0; i < 2; ++i)
#pragma unroll
      for (int j = 0; j < 2; ++j) {
        accg[i][j] = __builtin_amdgcn_wmma_f32_16x16x32_f16(
            false, afrag[i], false, bgf[j], (short)0, accg[i][j], false, false);
        accu[i][j] = __builtin_amdgcn_wmma_f32_16x16x32_f16(
            false, afrag[i], false, bufg[j], (short)0, accu[i][j], false, false);
      }

    if (has_next) {
      float g0v[4] = {g0.x, g0.y, g0.z, g0.w}, g1v[4] = {g1.x, g1.y, g1.z, g1.w};
      float u0v[4] = {u0.x, u0.y, u0.z, u0.w}, u1v[4] = {u1.x, u1.y, u1.z, u1.w};
#pragma unroll
      for (int j = 0; j < 4; ++j) {
        *(unsigned*)&Bg_nxt[(nc + j) * LDS_STRIDE + k2] = pack_h2(g0v[j], g1v[j]);
        *(unsigned*)&Bu_nxt[(nc + j) * LDS_STRIDE + k2] = pack_h2(u0v[j], u1v[j]);
      }
      wait_async0();
    }
    __syncthreads();
    buf ^= 1;
  }

  // epilogue: SwiGLU, store f16. C/D layout: VGPR r -> M = m0 + (lane<16?0:8) + r
  const int lane = tid & 31;
  const int nOff = lane & 15;
  const int mOff = (lane < 16) ? 0 : 8;
  _Float16* po = Out + e * strideOut;
#pragma unroll
  for (int i = 0; i < 2; ++i)
#pragma unroll
    for (int j = 0; j < 2; ++j) {
      long long m0 = mBase + wm + i * 16 + mOff;
      long long n0 = nBase + wn + j * 16 + nOff;
#pragma unroll
      for (int r = 0; r < 8; ++r) {
        float g = accg[i][j][r];
        float u = accu[i][j][r];
        float hv = g * fast_sigmoid(g) * u;
        po[(m0 + r) * (long long)N + n0] = (_Float16)hv;
      }
    }
}

// ---------------------------------------------------------------------------
// 5) GEMM with f32 output + optional fused per-row sigmoid gate (shared path)
// ---------------------------------------------------------------------------
__global__ __launch_bounds__(256)
void gemm_f32out_kernel(const _Float16* __restrict__ A, const float* __restrict__ B,
                        float* __restrict__ Out, int N, int Kdim,
                        long long strideA, long long strideB, long long strideOut,
                        const float* __restrict__ rowscale) {
  __shared__ __align__(16) _Float16 ldsA[2 * BUF_A];
  __shared__ __align__(16) _Float16 ldsB[2 * BUF_B];

  const int tid = threadIdx.x;
  const int e = blockIdx.z;
  const long long mBase = (long long)blockIdx.y * 128;
  const long long nBase = (long long)blockIdx.x * 64;
  const _Float16* Ae = A + e * strideA;
  const float*    Be = B + e * strideB;

  const int w  = tid >> 5;
  const int wm = (w & 3) * 32;
  const int wn = (w >> 2) * 32;

  v8f acc[2][2];
#pragma unroll
  for (int i = 0; i < 2; ++i)
#pragma unroll
    for (int j = 0; j < 2; ++j)
#pragma unroll
      for (int r = 0; r < 8; ++r) acc[i][j][r] = 0.f;

  const int ar = tid >> 1;
  const int ac = (tid & 1) * 16;
  const int k2 = (tid >> 4) * 2;
  const int nc = (tid & 15) * 4;

  // prologue: chunk 0 -> buffer 0
  {
    stage_a16(Ae + (mBase + ar) * (long long)Kdim + ac, &ldsA[ar * LDS_STRIDE + ac]);
    const float* pb = Be + (long long)k2 * N + nBase + nc;
    float4 b0 = *(const float4*)(pb);
    float4 b1 = *(const float4*)(pb + N);
    float b0v[4] = {b0.x, b0.y, b0.z, b0.w}, b1v[4] = {b1.x, b1.y, b1.z, b1.w};
#pragma unroll
    for (int j = 0; j < 4; ++j)
      *(unsigned*)&ldsB[(nc + j) * LDS_STRIDE + k2] = pack_h2(b0v[j], b1v[j]);
    wait_async0();
  }
  __syncthreads();

  int buf = 0;
  for (int kb = 0; kb < Kdim; kb += 32) {
    const bool has_next = (kb + 32) < Kdim;
    _Float16* A_cur = ldsA + buf * BUF_A;
    _Float16* B_cur = ldsB + buf * BUF_B;
    _Float16* A_nxt = ldsA + (buf ^ 1) * BUF_A;
    _Float16* B_nxt = ldsB + (buf ^ 1) * BUF_B;

    float4 b0{}, b1{};
    if (has_next) {
      stage_a16(Ae + (mBase + ar) * (long long)Kdim + kb + 32 + ac,
                &A_nxt[ar * LDS_STRIDE + ac]);
      const float* pb = Be + (long long)(kb + 32 + k2) * N + nBase + nc;
      b0 = *(const float4*)(pb);
      b1 = *(const float4*)(pb + N);
    }

    v16h afrag[2], bfrag[2];
#pragma unroll
    for (int i = 0; i < 2; ++i) afrag[i] = load_frag(A_cur, wm + i * 16, LDS_STRIDE);
#pragma unroll
    for (int j = 0; j < 2; ++j) bfrag[j] = load_frag(B_cur, wn + j * 16, LDS_STRIDE);
#pragma unroll
    for (int i = 0; i < 2; ++i)
#pragma unroll
      for (int j = 0; j < 2; ++j)
        acc[i][j] = __builtin_amdgcn_wmma_f32_16x16x32_f16(
            false, afrag[i], false, bfrag[j], (short)0, acc[i][j], false, false);

    if (has_next) {
      float b0v[4] = {b0.x, b0.y, b0.z, b0.w}, b1v[4] = {b1.x, b1.y, b1.z, b1.w};
#pragma unroll
      for (int j = 0; j < 4; ++j)
        *(unsigned*)&B_nxt[(nc + j) * LDS_STRIDE + k2] = pack_h2(b0v[j], b1v[j]);
      wait_async0();
    }
    __syncthreads();
    buf ^= 1;
  }

  const int lane = tid & 31;
  const int nOff = lane & 15;
  const int mOff = (lane < 16) ? 0 : 8;
  float* po = Out + e * strideOut;
#pragma unroll
  for (int i = 0; i < 2; ++i)
#pragma unroll
    for (int j = 0; j < 2; ++j) {
      long long m0 = mBase + wm + i * 16 + mOff;
      long long n0 = nBase + wn + j * 16 + nOff;
#pragma unroll
      for (int r = 0; r < 8; ++r) {
        float scl = rowscale ? rowscale[m0 + r] : 1.0f;
        po[(m0 + r) * (long long)N + n0] = acc[i][j][r] * scl;
      }
    }
}

// ---------------------------------------------------------------------------
// 6) Combine: per-token gather of its K expert rows (deterministic order)
// ---------------------------------------------------------------------------
__global__ __launch_bounds__(256)
void combine_kernel(const float* __restrict__ ye, const int* __restrict__ topi,
                    const float* __restrict__ topw, const int* __restrict__ pos,
                    float* __restrict__ out) {
  const int t = blockIdx.x;
  const int c0 = threadIdx.x * 8;
  float* po = out + (long long)t * H_DIM + c0;
  float acc[8];
#pragma unroll
  for (int j = 0; j < 8; ++j) acc[j] = po[j];
#pragma unroll
  for (int k = 0; k < 4; ++k) {
    int e = topi[t * 4 + k];
    int p = pos[t * 4 + k];
    float wv = topw[t * 4 + k];
    if (p < CAP_E) {
      const float* pe = ye + ((long long)e * CAP_E + p) * H_DIM + c0;
#pragma unroll
      for (int j = 0; j < 8; ++j) acc[j] = fmaf(wv, pe[j], acc[j]);
    }
  }
#pragma unroll
  for (int j = 0; j < 8; ++j) po[j] = acc[j];
}

// ---------------------------------------------------------------------------
extern "C" void kernel_launch(void* const* d_in, const int* in_sizes, int n_in,
                              void* d_out, int out_size, void* d_ws, size_t ws_size,
                              hipStream_t stream) {
  const float* x   = (const float*)d_in[0];
  const float* Wr  = (const float*)d_in[1];
  const float* Wg  = (const float*)d_in[2];
  const float* Wu  = (const float*)d_in[3];
  const float* Wd  = (const float*)d_in[4];
  const float* Wsg = (const float*)d_in[5];
  const float* Wsu = (const float*)d_in[6];
  const float* Wsd = (const float*)d_in[7];
  const float* Wse = (const float*)d_in[8];
  float* out = (float*)d_out;

  char* ws = (char*)d_ws;
  size_t off = 0;
  auto alloc = [&](size_t bytes) {
    size_t o = off;
    off = (off + bytes + 255) & ~(size_t)255;
    return o;
  };
  int*      topi   = (int*)     (ws + alloc((size_t)T_TOK * K_TOP * 4));
  float*    topw   = (float*)   (ws + alloc((size_t)T_TOK * K_TOP * 4));
  int*      pos    = (int*)     (ws + alloc((size_t)T_TOK * K_TOP * 4));
  int*      src    = (int*)     (ws + alloc((size_t)E_NUM * CAP_E * 4));
  float*    segate = (float*)   (ws + alloc((size_t)T_TOK * 4));
  _Float16* x_h    = (_Float16*)(ws + alloc((size_t)T_TOK * H_DIM * 2));
  _Float16* xe_h   = (_Float16*)(ws + alloc((size_t)E_NUM * CAP_E * H_DIM * 2)); // 32 MB
  _Float16* h1_h   = (_Float16*)(ws + alloc((size_t)E_NUM * CAP_E * F_DIM * 2)); // 23 MB
  float*    ye     = (float*)   (ws + alloc((size_t)E_NUM * CAP_E * H_DIM * 4)); // 64 MB
  // s1 (shared SwiGLU intermediate, 11.5 MB) reuses the xe region: xe is dead
  // after the expert SwiGLU GEMM, which completes before s1 is written.
  _Float16* s1_h   = xe_h;

  // 1) x -> f16
  cvt_f32_f16_kernel<<<(T_TOK * H_DIM) / (256 * 8), 256, 0, stream>>>(
      x, x_h, (long long)T_TOK * H_DIM);
  // 2) router (+ shared-expert sigmoid gate)
  router_kernel<<<T_TOK, 32, 0, stream>>>(x, Wr, Wse, topi, topw, segate);
  // 3) deterministic slot assignment + dispatch
  pos_kernel<<<1, 32, 0, stream>>>(topi, pos);
  init_src_kernel<<<(E_NUM * CAP_E) / 256, 256, 0, stream>>>(src);
  fill_src_kernel<<<(T_TOK * K_TOP) / 256, 256, 0, stream>>>(topi, pos, src);
  build_xe_kernel<<<E_NUM * CAP_E, 256, 0, stream>>>(x, src, xe_h);

  // 4) expert SwiGLU: h1 = silu(xe*Wg) * (xe*Wu)   [E,CAP,F]
  {
    dim3 grid(F_DIM / 64, CAP_E / 128, E_NUM);
    swiglu_gemm_kernel<<<grid, 256, 0, stream>>>(
        xe_h, Wg, Wu, h1_h, F_DIM, H_DIM,
        (long long)CAP_E * H_DIM, (long long)H_DIM * F_DIM, (long long)CAP_E * F_DIM);
  }
  // 5) shared SwiGLU: s1 = silu(x*Wsg) * (x*Wsu)   [T,FS]
  {
    dim3 grid(FS_DIM / 64, T_TOK / 128, 1);
    swiglu_gemm_kernel<<<grid, 256, 0, stream>>>(
        x_h, Wsg, Wsu, s1_h, FS_DIM, H_DIM, 0, 0, 0);
  }
  // 6) expert down: ye = h1 * Wd   [E,CAP,H]
  {
    dim3 grid(H_DIM / 64, CAP_E / 128, E_NUM);
    gemm_f32out_kernel<<<grid, 256, 0, stream>>>(
        h1_h, Wd, ye, H_DIM, F_DIM,
        (long long)CAP_E * F_DIM, (long long)F_DIM * H_DIM, (long long)CAP_E * H_DIM,
        nullptr);
  }
  // 7) shared down (+ sigmoid gate), writes every element of out
  {
    dim3 grid(H_DIM / 64, T_TOK / 128, 1);
    gemm_f32out_kernel<<<grid, 256, 0, stream>>>(
        s1_h, Wsd, out, H_DIM, FS_DIM, 0, 0, 0, segate);
  }
  // 8) combine expert outputs into out (per-token gather)
  combine_kernel<<<T_TOK, 256, 0, stream>>>(ye, topi, topw, pos, out);
}
